// BertMultiHeadAttention_78743930405233
// MI455X (gfx1250) — compile-verified, matrix operand order checked by reading, not run on hardware
//
#include <hip/hip_runtime.h>
#include <hip/hip_bf16.h>
#include <stdint.h>

// Problem constants (BERT MHA: B=4, S=1024, D=768, H=12, HD=64)
#define D_   768
#define H_   12
#define HD_  64
#define S_   1024
#define B_   4
#define M_   (B_ * S_)      // 4096 total rows
#define DD_  (D_ * D_)      // 589824
#define NEGV   (-10000.0f)
#define EPS_   (1e-12f)
#define SCALE_ (0.03608439182435161f)   // 1/sqrt(768)  (reference scales by sqrt(D), not sqrt(HD))

typedef __bf16 bf16;
typedef __attribute__((ext_vector_type(16))) __bf16 v16bf;
typedef __attribute__((ext_vector_type(8)))  __bf16 bf16x8;
typedef __attribute__((ext_vector_type(8)))  float  v8f;
typedef unsigned int u32x4 __attribute__((ext_vector_type(4)));
typedef int          i32x8 __attribute__((ext_vector_type(8)));
typedef int          i32x4 __attribute__((ext_vector_type(4)));

union Frag {
    v16bf  v;
    bf16x8 h[2];
    bf16   e[16];
};

__device__ inline v8f v8f_zero() {
    v8f z;
    #pragma unroll
    for (int i = 0; i < 8; ++i) z[i] = 0.0f;
    return z;
}

// D = A x B + C, bf16 inputs, f32 accumulate  -> v_wmma_f32_16x16x32_bf16
__device__ inline v8f wmma_bf16(v16bf a, v16bf b, v8f c) {
    return __builtin_amdgcn_wmma_f32_16x16x32_bf16(
        /*neg_a=*/false, a, /*neg_b=*/false, b,
        /*c_mod=*/(short)0, c, /*reuse_a=*/false, /*reuse_b=*/false);
}

// Load a 16x32 bf16 A/B fragment from a row-major [rows x ld] bf16 matrix.
// ISA layout: lane&15 selects row; koff = (lane>>4)*8;
// halves[0..7]  = K[k0+koff .. +7], halves[8..15] = K[k0+16+koff .. +7]
__device__ inline v16bf load_frag(const bf16* __restrict__ base, int row, int ld,
                                  int k0, int lane) {
    const int koff = (lane >> 4) * 8;
    const bf16* p = base + (size_t)row * ld + k0 + koff;
    Frag f;
    f.h[0] = *(const bf16x8*)(p);
    f.h[1] = *(const bf16x8*)(p + 16);
    return f.v;
}

// ---------------------------------------------------------------------------
// Tensor Data Mover: DMA a [rows x cols] bf16 tile (row stride `row_stride`
// elements) from global memory into LDS at byte offset `lds_off`.
// Builds the D# per CDNA5 ISA (groups 0/1, 2D tensor).
// Wave-level op; caller gates to one wave and follows with __syncthreads().
// ---------------------------------------------------------------------------
__device__ inline void tdm_load_2d(uint32_t lds_off, const bf16* __restrict__ gptr,
                                   int rows, int cols, int row_stride) {
#if __has_builtin(__builtin_amdgcn_tensor_load_to_lds)
    const uint64_t ga = (uint64_t)(uintptr_t)gptr;
    u32x4 g0;
    g0[0] = 1u;                                   // count=1, user descriptor
    g0[1] = lds_off;                              // lds_addr
    g0[2] = (uint32_t)(ga & 0xffffffffu);         // global_addr[31:0]
    g0[3] = (uint32_t)((ga >> 32) & 0x01ffffffu)  // global_addr[56:32]
          | (2u << 30);                           // type = 2 ("image")
    i32x8 g1;
    g1[0] = (int)(1u << 16);                                  // data_size=1 (2B)
    g1[1] = (int)(((uint32_t)cols & 0xffffu) << 16);          // tensor_dim0 lo16
    g1[2] = (int)((((uint32_t)cols >> 16) & 0xffffu)          // tensor_dim0 hi16
          | (((uint32_t)rows & 0xffffu) << 16));              // tensor_dim1 lo16
    g1[3] = (int)((((uint32_t)rows >> 16) & 0xffffu)          // tensor_dim1 hi16
          | (((uint32_t)cols & 0xffffu) << 16));              // tile_dim0
    g1[4] = (int)((uint32_t)rows & 0xffffu);                  // tile_dim1 (tile_dim2=0)
    g1[5] = row_stride;                                       // tensor_dim0_stride[31:0]
    g1[6] = 0;                                                // stride0 hi / stride1 lo
    g1[7] = 0;                                                // stride1 hi
    i32x4 z4;
    #pragma unroll
    for (int i = 0; i < 4; ++i) z4[i] = 0;
#if __clang_major__ >= 23
    i32x8 z8;
    #pragma unroll
    for (int i = 0; i < 8; ++i) z8[i] = 0;
    __builtin_amdgcn_tensor_load_to_lds(g0, g1, z4, z4, z8, 0);
#else
    __builtin_amdgcn_tensor_load_to_lds(g0, g1, z4, z4, 0);
#endif
    __builtin_amdgcn_s_wait_tensorcnt(0);
#else
    // Fallback: no TDM builtin available; this branch copies nothing.
    bf16* dst = (bf16*)(uintptr_t)0;  // placate unused warnings
    (void)dst;
    bf16* lds = (bf16*)((char*)nullptr + 0);
    (void)lds;
    // Plain copy via generic LDS pointer reconstructed by caller is not
    // possible here; the tile is not copied.
    // (This branch is not expected to be taken on probed toolchains.)
#endif
}

// ---------------------------------------------------------------------------
// Kernel 1: f32 -> bf16 conversion (grid-stride)
// ---------------------------------------------------------------------------
__global__ void cvt_f32_bf16(const float* __restrict__ src, bf16* __restrict__ dst, int n) {
    int i = blockIdx.x * blockDim.x + threadIdx.x;
    const int stride = gridDim.x * blockDim.x;
    for (; i < n; i += stride) dst[i] = (bf16)src[i];
}

// ---------------------------------------------------------------------------
// Kernel 2: fused QKV projection.  Out[z][m][n] = X[m][:] . Wz[n][:] + bz[n]
// grid = (768/64 = 12, 4096/128 = 32, 3), block = 256 (8 waves).
// Wave w owns rows [mbase + w*16, +16) x 64 output cols -> 4 WMMA n-tiles.
// ---------------------------------------------------------------------------
__global__ __launch_bounds__(256)
void qkv_gemm(const bf16* __restrict__ X, const bf16* __restrict__ W,
              const float* __restrict__ bq, const float* __restrict__ bk,
              const float* __restrict__ bv,
              bf16* __restrict__ Q, bf16* __restrict__ Ko, bf16* __restrict__ V) {
    const int lane  = threadIdx.x & 31;
    const int wave  = threadIdx.x >> 5;
    const int col   = lane & 15;
    const int rhalf = (lane >> 4) * 8;
    const int nbase = blockIdx.x * 64;
    const int mbase = blockIdx.y * 128 + wave * 16;
    const int z     = blockIdx.z;

    const bf16*  Wz   = W + (size_t)z * DD_;
    const float* bias = (z == 0) ? bq : (z == 1) ? bk : bv;
    bf16*        Out  = (z == 0) ? Q  : (z == 1) ? Ko : V;

    v8f acc[4];
    #pragma unroll
    for (int i = 0; i < 4; ++i) acc[i] = v8f_zero();

    const int arow = mbase + col;
    for (int k0 = 0; k0 < D_; k0 += 32) {
        const v16bf a = load_frag(X, arow, D_, k0, lane);
        #pragma unroll
        for (int nt = 0; nt < 4; ++nt) {
            const int n = nbase + nt * 16 + col;
            const v16bf bfr = load_frag(Wz, n, D_, k0, lane);
            acc[nt] = wmma_bf16(a, bfr, acc[nt]);
        }
    }
    #pragma unroll
    for (int nt = 0; nt < 4; ++nt) {
        const int n  = nbase + nt * 16 + col;
        const float bn = bias[n];
        #pragma unroll
        for (int r = 0; r < 8; ++r) {
            const int m = mbase + r + rhalf;
            Out[(size_t)m * D_ + n] = (bf16)(acc[nt][r] + bn);
        }
    }
}

// ---------------------------------------------------------------------------
// Kernel 3: fused attention per (b, h, 16-row q block).
//   Phase 1: scores = (Q Kt) * scale + masks, into 64KB LDS row buffer.
//            K chunks (128 keys x 64) staged into LDS by TDM.
//   Phase 2: row softmax (shuffle reductions), probs -> global output.
//   Phase 3: ctx = P V via WMMA (P converted bf16 from LDS, V via TDM).
// grid = (S/16 = 64, H = 12, B = 4), block = 256 (8 waves).
// ---------------------------------------------------------------------------
__global__ __launch_bounds__(256)
void attention(const bf16* __restrict__ Q, const bf16* __restrict__ K,
               const bf16* __restrict__ V, const float* __restrict__ pmask,
               float* __restrict__ probs, bf16* __restrict__ Ctx) {
    __shared__ float sc[16 * S_];       // 64 KB: 16 q-rows x 1024 scores
    __shared__ bf16  kvbuf[128 * HD_];  // 16 KB: staged K/V chunk

    const int lane  = threadIdx.x & 31;
    const int wave  = threadIdx.x >> 5;
    const int col   = lane & 15;
    const int rhalf = (lane >> 4) * 8;
    const int qbase = blockIdx.x * 16;
    const int h     = blockIdx.y;
    const int b     = blockIdx.z;

    // Per-wave Q fragments (16 rows x 64 hd) = two 16x32 A fragments.
    const int qrowg = b * S_ + qbase + col;
    const v16bf aq0 = load_frag(Q, qrowg, D_, h * HD_ +  0, lane);
    const v16bf aq1 = load_frag(Q, qrowg, D_, h * HD_ + 32, lane);

    const uint32_t kv_lds = (uint32_t)(uintptr_t)(void*)kvbuf;

    // ---- Phase 1: scores ----
    for (int c = 0; c < 8; ++c) {
        if (wave == 0) {
            tdm_load_2d(kv_lds, K + ((size_t)(b * S_ + c * 128)) * D_ + h * HD_,
                        128, HD_, D_);
            if (c + 1 < 8)   // warm L2 for the next chunk
                __builtin_prefetch(K + ((size_t)(b * S_ + (c + 1) * 128)) * D_ + h * HD_, 0, 1);
        }
        __syncthreads();

        v8f acc = v8f_zero();
        {
            const bf16* kb = kvbuf + (size_t)(wave * 16 + col) * HD_;
            Frag fb0, fb1;
            fb0.h[0] = *(const bf16x8*)(kb + rhalf);
            fb0.h[1] = *(const bf16x8*)(kb + rhalf + 16);
            fb1.h[0] = *(const bf16x8*)(kb + 32 + rhalf);
            fb1.h[1] = *(const bf16x8*)(kb + 32 + rhalf + 16);
            acc = wmma_bf16(aq0, fb0.v, acc);
            acc = wmma_bf16(aq1, fb1.v, acc);
        }
        const int   kcol = c * 128 + wave * 16 + col;
        const float pm   = pmask[b * S_ + kcol];
        const float addm = (1.0f - pm) * NEGV;
        #pragma unroll
        for (int r = 0; r < 8; ++r) {
            const int qr = qbase + r + rhalf;
            float v = acc[r] * SCALE_ + addm;
            if (kcol > qr) v = NEGV;   // causal masked_fill (set, not add)
            sc[(r + rhalf) * S_ + kcol] = v;
        }
        __syncthreads();
    }

    // ---- Phase 2: softmax over each of the 16 rows (16 lanes per row) ----
    {
        const int t = threadIdx.x, row = t >> 4, sub = t & 15;
        float mx = -3.4e38f;
        for (int j = sub; j < S_; j += 16) mx = fmaxf(mx, sc[row * S_ + j]);
        #pragma unroll
        for (int o = 8; o; o >>= 1) mx = fmaxf(mx, __shfl_xor(mx, o, 16));
        float sum = 0.0f;
        for (int j = sub; j < S_; j += 16) {
            const float p = __expf(sc[row * S_ + j] - mx);
            sc[row * S_ + j] = p;
            sum += p;
        }
        #pragma unroll
        for (int o = 8; o; o >>= 1) sum += __shfl_xor(sum, o, 16);
        const float inv = 1.0f / sum;
        float* prow = probs + (((size_t)(b * H_ + h)) * S_ + (qbase + row)) * S_;
        for (int j = sub; j < S_; j += 16) {
            const float p = sc[row * S_ + j] * inv;
            sc[row * S_ + j] = p;
            prow[j] = p;
        }
    }
    __syncthreads();

    // ---- Phase 3: ctx = P @ V  (waves 0..3 own the 4 output n-tiles) ----
    v8f cacc = v8f_zero();
    for (int c = 0; c < 8; ++c) {
        if (wave == 0)
            tdm_load_2d(kv_lds, V + ((size_t)(b * S_ + c * 128)) * D_ + h * HD_,
                        128, HD_, D_);
        __syncthreads();
        if (wave < 4) {
            #pragma unroll
            for (int kk = 0; kk < 4; ++kk) {
                const int kg = c * 128 + kk * 32;
                Frag fa, fb;
                #pragma unroll
                for (int j = 0; j < 8; ++j) {
                    fa.e[j]     = (bf16)sc[col * S_ + kg + rhalf + j];
                    fa.e[j + 8] = (bf16)sc[col * S_ + kg + rhalf + 16 + j];
                }
                #pragma unroll
                for (int j = 0; j < 8; ++j) {
                    fb.e[j]     = kvbuf[(size_t)(kk * 32 + rhalf + j)      * HD_ + wave * 16 + col];
                    fb.e[j + 8] = kvbuf[(size_t)(kk * 32 + rhalf + 16 + j) * HD_ + wave * 16 + col];
                }
                cacc = wmma_bf16(fa.v, fb.v, cacc);
            }
        }
        __syncthreads();
    }
    if (wave < 4) {
        #pragma unroll
        for (int r = 0; r < 8; ++r) {
            const int m    = b * S_ + qbase + r + rhalf;
            const int ncol = h * HD_ + wave * 16 + col;
            Ctx[(size_t)m * D_ + ncol] = (bf16)cacc[r];
        }
    }
}

// ---------------------------------------------------------------------------
// Kernel 4: out-proj + residual + LayerNorm over full 768-wide rows.
// grid = 4096/16 = 256, block = 256 (8 waves, 6 n-tiles each = 48 tiles).
// ---------------------------------------------------------------------------
__global__ __launch_bounds__(256)
void out_proj_ln(const bf16* __restrict__ Ctx, const bf16* __restrict__ Wo,
                 const float* __restrict__ bo, const float* __restrict__ emb,
                 const float* __restrict__ g, const float* __restrict__ bb,
                 float* __restrict__ out) {
    __shared__ float xbuf[16 * D_];   // 48 KB: residual-added rows
    const int lane  = threadIdx.x & 31;
    const int wave  = threadIdx.x >> 5;
    const int col   = lane & 15;
    const int rhalf = (lane >> 4) * 8;
    const int mbase = blockIdx.x * 16;
    const int arow  = mbase + col;

    v8f acc[6];
    #pragma unroll
    for (int i = 0; i < 6; ++i) acc[i] = v8f_zero();

    for (int k0 = 0; k0 < D_; k0 += 32) {
        const v16bf a = load_frag(Ctx, arow, D_, k0, lane);
        #pragma unroll
        for (int i = 0; i < 6; ++i) {
            const int n = (wave * 6 + i) * 16 + col;
            const v16bf bfr = load_frag(Wo, n, D_, k0, lane);
            acc[i] = wmma_bf16(a, bfr, acc[i]);
        }
    }
    #pragma unroll
    for (int i = 0; i < 6; ++i) {
        const int n = (wave * 6 + i) * 16 + col;
        const float bn = bo[n];
        #pragma unroll
        for (int r = 0; r < 8; ++r) {
            const int ml = r + rhalf;
            xbuf[ml * D_ + n] = acc[i][r] + bn + emb[(size_t)(mbase + ml) * D_ + n];
        }
    }
    __syncthreads();

    const int t = threadIdx.x, row = t >> 4, sub = t & 15;
    float s = 0.0f;
    for (int j = sub; j < D_; j += 16) s += xbuf[row * D_ + j];
    #pragma unroll
    for (int o = 8; o; o >>= 1) s += __shfl_xor(s, o, 16);
    const float mu = s * (1.0f / D_);
    float vs = 0.0f;
    for (int j = sub; j < D_; j += 16) {
        const float d = xbuf[row * D_ + j] - mu;
        vs += d * d;
    }
    #pragma unroll
    for (int o = 8; o; o >>= 1) vs += __shfl_xor(vs, o, 16);
    const float rstd = rsqrtf(vs * (1.0f / D_) + EPS_);
    for (int j = sub; j < D_; j += 16) {
        out[(size_t)(mbase + row) * D_ + j] =
            (xbuf[row * D_ + j] - mu) * rstd * g[j] + bb[j];
    }
}

// ---------------------------------------------------------------------------
// Host-side launcher
// ---------------------------------------------------------------------------
extern "C" void kernel_launch(void* const* d_in, const int* in_sizes, int n_in,
                              void* d_out, int out_size, void* d_ws, size_t ws_size,
                              hipStream_t stream) {
    (void)in_sizes; (void)n_in; (void)out_size; (void)ws_size;

    const float* emb   = (const float*)d_in[0];
    const float* pmask = (const float*)d_in[1];
    const float* wq    = (const float*)d_in[2];
    const float* bq    = (const float*)d_in[3];
    const float* wk    = (const float*)d_in[4];
    const float* bk    = (const float*)d_in[5];
    const float* wv    = (const float*)d_in[6];
    const float* bv    = (const float*)d_in[7];
    const float* wo    = (const float*)d_in[8];
    const float* bo    = (const float*)d_in[9];
    const float* ln_g  = (const float*)d_in[10];
    const float* ln_b  = (const float*)d_in[11];

    float* out_norm  = (float*)d_out;
    float* out_probs = out_norm + (size_t)M_ * D_;

    char* ws = (char*)d_ws;
    bf16* Xbf = (bf16*)ws;  ws += (size_t)M_ * D_ * 2;
    bf16* Wbf = (bf16*)ws;  ws += (size_t)4 * DD_ * 2;   // [q,k,v,o]
    bf16* Qb  = (bf16*)ws;  ws += (size_t)M_ * D_ * 2;
    bf16* Kb  = (bf16*)ws;  ws += (size_t)M_ * D_ * 2;
    bf16* Vb  = (bf16*)ws;  ws += (size_t)M_ * D_ * 2;
    bf16* Ctx = (bf16*)ws;                               // M_ * D_ bf16

    cvt_f32_bf16<<<2048, 256, 0, stream>>>(emb, Xbf, M_ * D_);
    cvt_f32_bf16<<<1024, 256, 0, stream>>>(wq, Wbf + 0 * (size_t)DD_, DD_);
    cvt_f32_bf16<<<1024, 256, 0, stream>>>(wk, Wbf + 1 * (size_t)DD_, DD_);
    cvt_f32_bf16<<<1024, 256, 0, stream>>>(wv, Wbf + 2 * (size_t)DD_, DD_);
    cvt_f32_bf16<<<1024, 256, 0, stream>>>(wo, Wbf + 3 * (size_t)DD_, DD_);

    qkv_gemm<<<dim3(12, 32, 3), 256, 0, stream>>>(Xbf, Wbf, bq, bk, bv, Qb, Kb, Vb);
    attention<<<dim3(64, 12, 4), 256, 0, stream>>>(Qb, Kb, Vb, pmask, out_probs, Ctx);
    out_proj_ln<<<256, 256, 0, stream>>>(Ctx, Wbf + 3 * (size_t)DD_, bo, emb,
                                         ln_g, ln_b, out_norm);
}
